// MultiLayerLSTM_53953379172842
// MI455X (gfx1250) — compile-verified, matrix operand order checked by reading, not run on hardware
//
#include <hip/hip_runtime.h>
#include <hip/hip_bf16.h>

// ---------------------------------------------------------------------------
// Multi-layer LSTM for MI455X (gfx1250), compile-only target.
// Persistent wavefront-pipelined kernel: 64 blocks (32 per layer), bf16 WMMA
// (v_wmma_f32_16x16x32_bf16), LDS-resident weight slices, monitor-based grid
// barrier per time step, double-buffered h state, c state resident in LDS,
// global_prefetch for the streamed x operand, NT stores for the f32 output.
// ---------------------------------------------------------------------------

#define LSTM_B 32
#define LSTM_T 2048
#define LSTM_I 256
#define LSTM_H 512
#define LSTM_G (4 * LSTM_H)   // 2048 gate columns
#define NBLK_PER_LAYER 32     // each owns 16 h-columns -> 4x16 gate columns
#define NBLK_TOTAL 64

typedef __attribute__((ext_vector_type(16))) __bf16 v16bf;
typedef __attribute__((ext_vector_type(8)))  float  v8f;

__device__ __forceinline__ unsigned short f2bf(float f) {
    unsigned int u = __float_as_uint(f);
    unsigned int r = u + 0x7FFFu + ((u >> 16) & 1u);  // round-to-nearest-even
    return (unsigned short)(r >> 16);
}

// Load one 16-lane-striped bf16 WMMA fragment (A or B operand).
// p must already point at: rowBase + k0 + (lane>=16 ? 8 : 0).
// Layout per ISA 7.12.2: 8 contiguous bf16 at +0 and 8 more at +16.
__device__ __forceinline__ v16bf ld_frag(const unsigned short* p) {
    union { int4 q[2]; v16bf v; } u;
    u.q[0] = *(const int4*)(p);
    u.q[1] = *(const int4*)(p + 16);
    return u.v;
}

__device__ __forceinline__ v8f wmma_bf16(v16bf a, v16bf b, v8f c) {
    return __builtin_amdgcn_wmma_f32_16x16x32_bf16(
        /*neg_a=*/false, a, /*neg_b=*/false, b,
        /*c_mod=*/(short)0, c, /*reuse_a=*/false, /*reuse_b=*/false);
}

__device__ __forceinline__ float sigmoidf_(float x) {
    return 1.0f / (1.0f + __expf(-x));
}

// Monitor-armed load of the barrier generation counter: asks GL2 to wake this
// wave when the cacheline is next written (CDNA5 MWAIT path, ISA ch.10 §6).
__device__ __forceinline__ int monitor_load_i32(const int* p) {
    int v;
    asm volatile("global_load_monitor_b32 %0, %1, off\n\t"
                 "s_wait_loadcnt 0x0"
                 : "=v"(v) : "v"(p) : "memory");
    return v;
}

// Sense-reversing grid barrier over NBLK_TOTAL co-resident blocks.
// Spinners sleep on the monitor instead of polling: the releasing
// atomicAdd(gen) dirties the watched line and wakes them.
__device__ __forceinline__ void grid_barrier(int* cnt, volatile int* gen) {
    __syncthreads();
    if (threadIdx.x == 0) {
        __threadfence();
        int g = *gen;
        if (atomicAdd(cnt, 1) == NBLK_TOTAL - 1) {
            *cnt = 0;
            __threadfence();
            atomicAdd((int*)gen, 1);
        } else {
            for (;;) {
                int v = monitor_load_i32((const int*)gen);
                if (v != g) break;
                asm volatile("s_monitor_sleep 1" ::: "memory");
            }
        }
    }
    __syncthreads();
}

// ------------------------------ prep kernels -------------------------------

__global__ void lstm_cvt_bf16(const float* __restrict__ in,
                              unsigned short* __restrict__ out, size_t n) {
    size_t i = (size_t)blockIdx.x * blockDim.x + threadIdx.x;
    if (i < n) out[i] = f2bf(in[i]);
}

// W: [K][N] f32 row-major  ->  WT: [N][K] bf16 (so per-lane B loads are contiguous)
__global__ void lstm_transpose_cvt(const float* __restrict__ W,
                                   unsigned short* __restrict__ WT,
                                   int K, int N) {
    size_t i = (size_t)blockIdx.x * blockDim.x + threadIdx.x;
    if (i < (size_t)K * N) {
        int n = (int)(i / K);
        int k = (int)(i % K);
        WT[i] = f2bf(W[(size_t)k * N + n]);
    }
}

__global__ void lstm_init_state(const float* __restrict__ h0,
                                unsigned short* __restrict__ hbuf0,
                                unsigned short* __restrict__ hbuf1,
                                int* __restrict__ bars) {
    int i = blockIdx.x * blockDim.x + threadIdx.x;
    if (i < LSTM_B * LSTM_H) {
        hbuf0[i] = f2bf(h0[i]);                       // layer 0, parity 0
        hbuf1[i] = f2bf(h0[LSTM_B * LSTM_H + i]);     // layer 1, parity 0
    }
    if (i < 4) bars[i] = 0;
}

// ---------------------------- persistent kernel ----------------------------
// grid = 64 blocks x 256 threads (8 wave32).
// blockIdx/32 -> layer; blockIdx%32 -> 16-wide h-column slice j.
// wave w: m_tile = w&1 (batch rows 0-15 / 16-31), gate = w>>1 (i,f,g,o).

__global__ void lstm_persistent(
    const unsigned short* __restrict__ xbf,     // [B][T][I] bf16 (layer0 input)
    unsigned short* __restrict__ seq0bf,        // [B][T][H] bf16 (layer0 out / layer1 in)
    const unsigned short* __restrict__ wihT0,   // [G][I]  bf16
    const unsigned short* __restrict__ whhT0,   // [G][H]  bf16
    const float* __restrict__ b0,               // [G]
    const unsigned short* __restrict__ wihT1,   // [G][H]  bf16
    const unsigned short* __restrict__ whhT1,   // [G][H]  bf16
    const float* __restrict__ b1,               // [G]
    unsigned short* __restrict__ hbuf0,         // [2][B][H] bf16
    unsigned short* __restrict__ hbuf1,         // [2][B][H] bf16
    const float* __restrict__ c0_init,          // [L][B][H] f32
    float* __restrict__ out_seq,                // [B][T][H] f32
    float* __restrict__ out_h,                  // [B][H] f32
    float* __restrict__ out_c,                  // [B][H] f32
    int* __restrict__ bar_cnt,
    int* __restrict__ bar_gen) {

    const int layer = blockIdx.x >> 5;
    const int j     = blockIdx.x & 31;          // h-column slice [16j, 16j+16)
    const int tid   = threadIdx.x;
    const int lane  = tid & 31;
    const int wave  = tid >> 5;
    const int m_tile = wave & 1;
    const int gate   = wave >> 1;               // 0:i 1:f 2:g 3:o

    const int Kx = layer ? LSTM_H : LSTM_I;
    const unsigned short* wihT = layer ? wihT1 : wihT0;
    const unsigned short* whhT = layer ? whhT1 : whhT0;
    const float* bias          = layer ? b1 : b0;
    const unsigned short* xsrc = layer ? seq0bf : xbf;
    const size_t x_row_stride  = (size_t)LSTM_T * Kx;   // stride between batch rows
    unsigned short* hbuf       = layer ? hbuf1 : hbuf0;

    extern __shared__ char smem[];
    unsigned short* s_wih = (unsigned short*)smem;                       // [64][Kx]
    unsigned short* s_whh = (unsigned short*)(smem + (size_t)64 * Kx * 2); // [64][H]
    float* s_g = (float*)(smem + (size_t)64 * Kx * 2 + (size_t)64 * LSTM_H * 2); // [4][32][16]
    float* s_c = s_g + 4 * 32 * 16;                                      // [32][16]

    // ---- load weight slices into LDS (once) -------------------------------
    {
        const int kq_ih = Kx >> 3;              // int4s (8 bf16) per row
        for (int idx = tid; idx < 64 * kq_ih; idx += 256) {
            int cl = idx / kq_ih, kq = idx % kq_ih;
            int c  = (cl >> 4) * LSTM_H + (j << 4) + (cl & 15);   // global gate col
            ((int4*)s_wih)[(size_t)cl * kq_ih + kq] =
                ((const int4*)(wihT + (size_t)c * Kx))[kq];
        }
        const int kq_hh = LSTM_H >> 3;
        for (int idx = tid; idx < 64 * kq_hh; idx += 256) {
            int cl = idx / kq_hh, kq = idx % kq_hh;
            int c  = (cl >> 4) * LSTM_H + (j << 4) + (cl & 15);
            ((int4*)s_whh)[(size_t)cl * kq_hh + kq] =
                ((const int4*)(whhT + (size_t)c * LSTM_H))[kq];
        }
        for (int idx = tid; idx < 32 * 16; idx += 256) {
            int m = idx >> 4, q = idx & 15;
            s_c[idx] = c0_init[(size_t)layer * LSTM_B * LSTM_H +
                               (size_t)m * LSTM_H + (j << 4) + q];
        }
    }
    __syncthreads();

    const int rowm = lane & 15;        // A-fragment row within tile
    const int hsel = lane >> 4;        // 0: K {0..7,16..23}, 1: K {8..15,24..31}
    const int nl   = lane & 15;        // B-fragment / C-tile column
    const int m0   = m_tile << 4;
    const float bv = bias[gate * LSTM_H + (j << 4) + nl];

    const unsigned short* b_ih = s_wih + (size_t)((gate << 4) + nl) * Kx + hsel * 8;
    const unsigned short* b_hh = s_whh + (size_t)((gate << 4) + nl) * LSTM_H + hsel * 8;
    const unsigned short* a_row_base = xsrc + (size_t)(m0 + rowm) * x_row_stride;

    // ---- wavefront-pipelined time loop ------------------------------------
    for (int s = 0; s < LSTM_T + 1; ++s) {
        const bool active = layer == 0 ? (s < LSTM_T) : (s >= 1);
        const int  t      = layer == 0 ? s : (s - 1);

        if (active) {
            // Prefetch next step's streamed x rows into cache (layer 0 only:
            // layer 1's input was produced one barrier ago and is L2-hot).
            if (layer == 0 && t + 1 < LSTM_T) {
                const unsigned short* nx = a_row_base + (size_t)(t + 1) * Kx;
                for (int o = 0; o < Kx; o += 64)            // one line / 128B
                    __builtin_prefetch(nx + o, 0, 3);
            }

            v8f acc0 = {bv, bv, bv, bv, bv, bv, bv, bv};
            v8f acc1 = {};

            // x_t @ W_ih slice  (K = Kx)
            const unsigned short* a_x = a_row_base + (size_t)t * Kx + hsel * 8;
            const int nkx = Kx >> 5;
#pragma unroll 4
            for (int kk = 0; kk < nkx; ++kk) {
                v16bf a = ld_frag(a_x + (kk << 5));
                v16bf b = ld_frag(b_ih + (kk << 5));
                if (kk & 1) acc1 = wmma_bf16(a, b, acc1);
                else        acc0 = wmma_bf16(a, b, acc0);
            }
            // h_t @ W_hh slice  (K = H)
            const unsigned short* a_h = hbuf + (size_t)(t & 1) * LSTM_B * LSTM_H +
                (size_t)(m0 + rowm) * LSTM_H + hsel * 8;
#pragma unroll 4
            for (int kk = 0; kk < (LSTM_H >> 5); ++kk) {
                v16bf a = ld_frag(a_h + (kk << 5));
                v16bf b = ld_frag(b_hh + (kk << 5));
                if (kk & 1) acc1 = wmma_bf16(a, b, acc1);
                else        acc0 = wmma_bf16(a, b, acc0);
            }
            // spill gate tile: C layout row = r + 8*hsel, col = nl
#pragma unroll
            for (int r = 0; r < 8; ++r) {
                int m = m0 + r + (hsel << 3);
                s_g[(size_t)(gate * 32 + m) * 16 + nl] = acc0[r] + acc1[r];
            }
        }
        __syncthreads();

        if (active) {
            // fused LSTM cell update over this block's [32 x 16] slice
            for (int e = tid; e < 32 * 16; e += 256) {
                int m = e >> 4, q = e & 15;
                float gi = s_g[(size_t)(0 * 32 + m) * 16 + q];
                float gf = s_g[(size_t)(1 * 32 + m) * 16 + q];
                float gg = s_g[(size_t)(2 * 32 + m) * 16 + q];
                float go = s_g[(size_t)(3 * 32 + m) * 16 + q];
                float iv = sigmoidf_(gi);
                float fv = sigmoidf_(gf);
                float gv = tanhf(gg);
                float ov = sigmoidf_(go);
                float cn = fv * s_c[e] + iv * gv;
                float hn = ov * tanhf(cn);
                s_c[e] = cn;
                unsigned short hb = f2bf(hn);
                hbuf[(size_t)((t + 1) & 1) * LSTM_B * LSTM_H +
                     (size_t)m * LSTM_H + (j << 4) + q] = hb;
                if (layer == 0) {
                    seq0bf[(size_t)m * LSTM_T * LSTM_H + (size_t)t * LSTM_H +
                           (j << 4) + q] = hb;
                } else {
                    // streamed, never re-read -> non-temporal store
                    __builtin_nontemporal_store(
                        hn, &out_seq[(size_t)m * LSTM_T * LSTM_H +
                                     (size_t)t * LSTM_H + (j << 4) + q]);
                    if (t == LSTM_T - 1) {
                        __builtin_nontemporal_store(
                            hn, &out_h[(size_t)m * LSTM_H + (j << 4) + q]);
                        __builtin_nontemporal_store(
                            cn, &out_c[(size_t)m * LSTM_H + (j << 4) + q]);
                    }
                }
            }
        }
        grid_barrier(bar_cnt, bar_gen);
    }
}

// ------------------------------- launcher ----------------------------------

extern "C" void kernel_launch(void* const* d_in, const int* in_sizes, int n_in,
                              void* d_out, int out_size, void* d_ws, size_t ws_size,
                              hipStream_t stream) {
    const float* x    = (const float*)d_in[0];   // [B][T][I]
    const float* h0   = (const float*)d_in[1];   // [L][B][H]
    const float* c0   = (const float*)d_in[2];   // [L][B][H]
    const float* Wih0 = (const float*)d_in[3];   // [I][4H]
    const float* Whh0 = (const float*)d_in[4];   // [H][4H]
    const float* b0   = (const float*)d_in[5];   // [4H]
    const float* Wih1 = (const float*)d_in[6];   // [H][4H]
    const float* Whh1 = (const float*)d_in[7];   // [H][4H]
    const float* b1   = (const float*)d_in[8];   // [4H]

    float* out_seq = (float*)d_out;
    float* out_h   = out_seq + (size_t)LSTM_B * LSTM_T * LSTM_H;
    float* out_c   = out_h + (size_t)LSTM_B * LSTM_H;

    // workspace carve-up (256B aligned)
    char* ws = (char*)d_ws;
    size_t off = 0;
    auto carve = [&](size_t bytes) {
        char* p = ws + off;
        off += (bytes + 255) & ~(size_t)255;
        return p;
    };
    unsigned short* xbf    = (unsigned short*)carve((size_t)LSTM_B * LSTM_T * LSTM_I * 2);
    unsigned short* seq0bf = (unsigned short*)carve((size_t)LSTM_B * LSTM_T * LSTM_H * 2);
    unsigned short* wihT0  = (unsigned short*)carve((size_t)LSTM_G * LSTM_I * 2);
    unsigned short* whhT0  = (unsigned short*)carve((size_t)LSTM_G * LSTM_H * 2);
    unsigned short* wihT1  = (unsigned short*)carve((size_t)LSTM_G * LSTM_H * 2);
    unsigned short* whhT1  = (unsigned short*)carve((size_t)LSTM_G * LSTM_H * 2);
    unsigned short* hbuf0  = (unsigned short*)carve((size_t)2 * LSTM_B * LSTM_H * 2);
    unsigned short* hbuf1  = (unsigned short*)carve((size_t)2 * LSTM_B * LSTM_H * 2);
    int*            bars   = (int*)carve(256);

    // 1) input -> bf16
    {
        size_t n = (size_t)LSTM_B * LSTM_T * LSTM_I;
        lstm_cvt_bf16<<<(unsigned)((n + 255) / 256), 256, 0, stream>>>(x, xbf, n);
    }
    // 2) weights -> transposed bf16 [N][K]
    {
        size_t n0 = (size_t)LSTM_I * LSTM_G;
        lstm_transpose_cvt<<<(unsigned)((n0 + 255) / 256), 256, 0, stream>>>(
            Wih0, wihT0, LSTM_I, LSTM_G);
        size_t n1 = (size_t)LSTM_H * LSTM_G;
        lstm_transpose_cvt<<<(unsigned)((n1 + 255) / 256), 256, 0, stream>>>(
            Whh0, whhT0, LSTM_H, LSTM_G);
        lstm_transpose_cvt<<<(unsigned)((n1 + 255) / 256), 256, 0, stream>>>(
            Wih1, wihT1, LSTM_H, LSTM_G);
        lstm_transpose_cvt<<<(unsigned)((n1 + 255) / 256), 256, 0, stream>>>(
            Whh1, whhT1, LSTM_H, LSTM_G);
    }
    // 3) h state init + barrier counters
    {
        int n = LSTM_B * LSTM_H;
        lstm_init_state<<<(n + 255) / 256, 256, 0, stream>>>(h0, hbuf0, hbuf1, bars);
    }
    // 4) persistent wavefront-pipelined recurrence
    {
        // worst case (layer1): 64*H*2 + 64*H*2 + 8K gate tiles + 2K c state
        const size_t shmem = (size_t)64 * LSTM_H * 2 + (size_t)64 * LSTM_H * 2 +
                             4 * 32 * 16 * 4 + 32 * 16 * 4;
        (void)hipFuncSetAttribute((const void*)lstm_persistent,
                                  hipFuncAttributeMaxDynamicSharedMemorySize,
                                  (int)shmem);
        lstm_persistent<<<NBLK_TOTAL, 256, shmem, stream>>>(
            xbf, seq0bf, wihT0, whhT0, b0, wihT1, whhT1, b1,
            hbuf0, hbuf1, c0, out_seq, out_h, out_c, bars, bars + 1);
    }
}
